// GraphAttentionLayer_60498909331487
// MI455X (gfx1250) — compile-verified, matrix operand order checked by reading, not run on hardware
//
#include <hip/hip_runtime.h>

// ---------------- CDNA5 WMMA types ----------------
typedef __attribute__((ext_vector_type(16))) __bf16 v16bf;
typedef __attribute__((ext_vector_type(8)))  float  v8f;

union U16 { v16bf v; unsigned int u[8]; uint4 q[2]; };

__device__ __forceinline__ unsigned short f2bf(float f) {
    // round-to-nearest-even fp32 -> bf16
    unsigned int x = __float_as_uint(f);
    unsigned int r = x + 0x7FFFu + ((x >> 16) & 1u);
    return (unsigned short)(r >> 16);
}

// Load one 16-bit A/B fragment (ISA 7.12.2 layout) as two 16-byte vectors.
// base = start of this lane's row (A: row m = lane&15; B: col n = lane&15).
// VGPR p=0..3 hold K pairs (kt*32 + hi*8 + 0,2,4,6), p=4..7 hold +16.
__device__ __forceinline__ void load_frag(U16& f, const unsigned short* base, int kt, int hi) {
    const unsigned short* p0 = base + kt * 32 + hi * 8;
    f.q[0] = *(const uint4*)(p0);
    f.q[1] = *(const uint4*)(p0 + 16);
}

__device__ __forceinline__ float redux16(float x) {
    // sum across the 16-lane half-group (xor masks stay within the half)
    x += __shfl_xor(x, 1, 32);
    x += __shfl_xor(x, 2, 32);
    x += __shfl_xor(x, 4, 32);
    x += __shfl_xor(x, 8, 32);
    return x;
}

constexpr int N_ATOMS = 100000;
constexpr int M_NBR   = 12;
constexpr int F_DIM   = 128;   // H*D = 8*16
constexpr int NF_DIM  = 64;
constexpr int N_TILES = N_ATOMS / 16;   // 6250 exactly

// ---------------- Kernel 0: fp32 weights -> bf16 transposed [n][k] ----------------
__global__ void prep_weights_kernel(const float* __restrict__ Wq, const float* __restrict__ Wk,
                                    const float* __restrict__ Wv, const float* __restrict__ Wo,
                                    const float* __restrict__ Wn, unsigned short* __restrict__ wbf) {
    int tid = blockIdx.x * blockDim.x + threadIdx.x;
    int stride = gridDim.x * blockDim.x;
    for (int i = tid; i < 128 * 128; i += stride) {
        int n = i >> 7, k = i & 127;
        wbf[0 * 16384 + i] = f2bf(Wq[k * 128 + n]);
        wbf[1 * 16384 + i] = f2bf(Wk[k * 128 + n]);
        wbf[2 * 16384 + i] = f2bf(Wv[k * 128 + n]);
        wbf[3 * 16384 + i] = f2bf(Wo[k * 128 + n]);
    }
    for (int i = tid; i < 128 * 64; i += stride) {
        int n = i >> 6, k = i & 63;
        wbf[4 * 16384 + i] = f2bf(Wn[k * 128 + n]);
    }
}

// ---------------- Kernel 1: q/k/v = atom_fea @ W + b  (bf16 WMMA, f32 acc) ----------------
__global__ __launch_bounds__(32) void qkv_kernel(
    const float* __restrict__ atom,
    const float* __restrict__ bq, const float* __restrict__ bk, const float* __restrict__ bv,
    const unsigned short* __restrict__ wqt, const unsigned short* __restrict__ wkt,
    const unsigned short* __restrict__ wvt,
    float* __restrict__ q, float* __restrict__ k, float* __restrict__ v) {

    __shared__ unsigned short sA[16 * 128];
    const int a0   = blockIdx.x << 4;
    const int lane = threadIdx.x;
    const int hi   = lane >> 4;
    const int lo   = lane & 15;

    // stage 16x128 atom tile as bf16 in LDS
    {
        int row = lane >> 1;
        int c0  = (lane & 1) * 64;
        const float* src = atom + (size_t)(a0 + row) * F_DIM + c0;
        #pragma unroll
        for (int j = 0; j < 64; j += 4) {
            float4 f = *(const float4*)(src + j);
            sA[row * 128 + c0 + j + 0] = f2bf(f.x);
            sA[row * 128 + c0 + j + 1] = f2bf(f.y);
            sA[row * 128 + c0 + j + 2] = f2bf(f.z);
            sA[row * 128 + c0 + j + 3] = f2bf(f.w);
        }
    }
    __syncthreads();

    // A fragments for K = 0..127 (4 k-tiles), 2 x ds_load_b128 each
    U16 afrag[4];
    #pragma unroll
    for (int kt = 0; kt < 4; ++kt) load_frag(afrag[kt], &sA[lo * 128], kt, hi);

    const unsigned short* wts[3] = {wqt, wkt, wvt};
    const float* bias[3]         = {bq, bk, bv};
    float* outs[3]               = {q, k, v};

    #pragma unroll
    for (int mtx = 0; mtx < 3; ++mtx) {
        const unsigned short* wp = wts[mtx] + lo * 128;   // + ntile*16*128 below
        float* op = outs[mtx];
        #pragma unroll
        for (int nt = 0; nt < 8; ++nt) {
            v8f acc = {};
            #pragma unroll
            for (int kt = 0; kt < 4; ++kt) {
                U16 bfrag;
                load_frag(bfrag, wp + nt * 16 * 128, kt, hi);
                acc = __builtin_amdgcn_wmma_f32_16x16x32_bf16(
                    false, afrag[kt].v, false, bfrag.v, (short)0, acc, false, false);
            }
            float b = bias[mtx][nt * 16 + lo];
            #pragma unroll
            for (int r = 0; r < 8; ++r) {
                // independent address per row so stores can clause
                float* dst = op + (size_t)(a0 + r + hi * 8) * F_DIM + nt * 16 + lo;
                *dst = acc[r] + b;
            }
        }
    }
}

// ---------------- Kernel 2: fused neighbor-proj + attention + Wo + gate + LN ----------------
__global__ __launch_bounds__(32) void attn_kernel(
    const float* __restrict__ atom, const float* __restrict__ nbr_fea,
    const long long* __restrict__ nbr_idx,
    const unsigned short* __restrict__ wnt, const unsigned short* __restrict__ wot,
    const float* __restrict__ bn, const float* __restrict__ bo,
    const float* __restrict__ Wg, const float* __restrict__ bg,
    const float* __restrict__ gamma, const float* __restrict__ beta,
    const float* __restrict__ q, const float* __restrict__ k, const float* __restrict__ v,
    float* __restrict__ out) {

    __shared__ unsigned short sNb[16 * 64];    // nbr_fea tile (bf16)
    __shared__ float          sK[16 * 128];    // gathered k rows
    __shared__ float          sV[16 * 128];    // gathered v rows
    __shared__ unsigned short sAtt[16 * 128];  // att tile (bf16) for Wo GEMM

    const int a0   = blockIdx.x << 4;
    const int lane = threadIdx.x;
    const int hi   = lane >> 4;
    const int lo   = lane & 15;

    // preload Wn B-fragments (reused across all 12 slots): 8 heads x 2 k-tiles
    U16 wnb[8][2];
    #pragma unroll
    for (int h = 0; h < 8; ++h)
        #pragma unroll
        for (int kt = 0; kt < 2; ++kt)
            load_frag(wnb[h][kt], wnt + (h * 16 + lo) * 64, kt, hi);

    // preload q tile in D-fragment layout: element (m = r+8*hi, col = h*16+lo)
    float qf[8][8];
    #pragma unroll
    for (int h = 0; h < 8; ++h)
        #pragma unroll
        for (int r = 0; r < 8; ++r)
            qf[h][r] = q[(size_t)(a0 + r + hi * 8) * F_DIM + h * 16 + lo];

    float attacc[8][8], rmax[8][8], rsum[8][8];
    #pragma unroll
    for (int h = 0; h < 8; ++h)
        #pragma unroll
        for (int r = 0; r < 8; ++r) { attacc[h][r] = 0.f; rmax[h][r] = -3.0e38f; rsum[h][r] = 0.f; }

    for (int s = 0; s < M_NBR; ++s) {
        // stage 16x64 neighbor-feature tile (bf16)
        {
            int row = lane >> 1;
            int c0  = (lane & 1) * 32;
            const float* src = nbr_fea + ((size_t)(a0 + row) * M_NBR + s) * NF_DIM + c0;
            #pragma unroll
            for (int j = 0; j < 32; j += 4) {
                float4 f = *(const float4*)(src + j);
                sNb[row * 64 + c0 + j + 0] = f2bf(f.x);
                sNb[row * 64 + c0 + j + 1] = f2bf(f.y);
                sNb[row * 64 + c0 + j + 2] = f2bf(f.z);
                sNb[row * 64 + c0 + j + 3] = f2bf(f.w);
            }
        }
        // gather k/v rows for this slot
        {
            int row = lane >> 1;
            int c0  = (lane & 1) * 64;
            long long gi = nbr_idx[(size_t)(a0 + row) * M_NBR + s];
            const float* ks = k + (size_t)gi * F_DIM + c0;
            const float* vs = v + (size_t)gi * F_DIM + c0;
            #pragma unroll
            for (int j = 0; j < 64; j += 4) {
                *(float4*)(&sK[row * 128 + c0 + j]) = *(const float4*)(ks + j);
                *(float4*)(&sV[row * 128 + c0 + j]) = *(const float4*)(vs + j);
            }
        }
        __syncthreads();

        // A fragments of the 16x64 neighbor tile
        U16 na[2];
        #pragma unroll
        for (int kt = 0; kt < 2; ++kt) load_frag(na[kt], &sNb[lo * 64], kt, hi);

        #pragma unroll
        for (int h = 0; h < 8; ++h) {
            v8f acc = {};
            acc = __builtin_amdgcn_wmma_f32_16x16x32_bf16(false, na[0].v, false, wnb[h][0].v,
                                                          (short)0, acc, false, false);
            acc = __builtin_amdgcn_wmma_f32_16x16x32_bf16(false, na[1].v, false, wnb[h][1].v,
                                                          (short)0, acc, false, false);
            float bnv = bn[h * 16 + lo];
            #pragma unroll
            for (int r = 0; r < 8; ++r) {
                int m = r + hi * 8;
                float nt  = acc[r] + bnv;                      // nbr_t element
                float nkv = nt + sK[m * 128 + h * 16 + lo];    // nk
                float nvv = nt + sV[m * 128 + h * 16 + lo];    // nv
                float sc  = redux16(qf[h][r] * nkv) * 0.25f;   // / sqrt(D=16); uniform in half-group
                float mo  = rmax[h][r];
                float mn  = fmaxf(mo, sc);
                float cor = __expf(mo - mn);
                float pw  = __expf(sc - mn);
                rsum[h][r]   = rsum[h][r] * cor + pw;
                attacc[h][r] = attacc[h][r] * cor + pw * nvv;
                rmax[h][r]   = mn;
            }
        }
        __syncthreads();   // protect LDS before next slot overwrites it
    }

    // finalize softmax-weighted V, stage as bf16 A tile for Wo GEMM
    #pragma unroll
    for (int h = 0; h < 8; ++h)
        #pragma unroll
        for (int r = 0; r < 8; ++r) {
            attacc[h][r] /= rsum[h][r];
            int m = r + hi * 8;
            sAtt[m * 128 + h * 16 + lo] = f2bf(attacc[h][r]);
        }
    __syncthreads();

    U16 aa[4];
    #pragma unroll
    for (int kt = 0; kt < 4; ++kt) load_frag(aa[kt], &sAtt[lo * 128], kt, hi);

    float o[8][8];
    #pragma unroll
    for (int nt = 0; nt < 8; ++nt) {
        v8f acc = {};
        const unsigned short* wp = wot + (nt * 16 + lo) * 128;
        #pragma unroll
        for (int kt = 0; kt < 4; ++kt) {
            U16 bf;
            load_frag(bf, wp, kt, hi);
            acc = __builtin_amdgcn_wmma_f32_16x16x32_bf16(false, aa[kt].v, false, bf.v,
                                                          (short)0, acc, false, false);
        }
        float b = bo[nt * 16 + lo];
        #pragma unroll
        for (int r = 0; r < 8; ++r) o[nt][r] = acc[r] + b;
    }

    // residual (fp32)
    float res[8][8];
    #pragma unroll
    for (int h = 0; h < 8; ++h)
        #pragma unroll
        for (int r = 0; r < 8; ++r)
            res[h][r] = atom[(size_t)(a0 + r + hi * 8) * F_DIM + h * 16 + lo];

    // gate: sigmoid([out, residual] @ Wg + bg)
    float wg0[8], wg1[8];
    #pragma unroll
    for (int h = 0; h < 8; ++h) { wg0[h] = Wg[h * 16 + lo]; wg1[h] = Wg[128 + h * 16 + lo]; }
    float bgv = bg[0];

    #pragma unroll
    for (int r = 0; r < 8; ++r) {
        float part = 0.f;
        #pragma unroll
        for (int h = 0; h < 8; ++h) part += o[h][r] * wg0[h] + res[h][r] * wg1[h];
        part = redux16(part);
        float g = 1.f / (1.f + __expf(-(part + bgv)));
        #pragma unroll
        for (int h = 0; h < 8; ++h) o[h][r] = g * o[h][r] + (1.f - g) * res[h][r];
    }

    // layernorm over 128 features per row, then store
    #pragma unroll
    for (int r = 0; r < 8; ++r) {
        float sm = 0.f;
        #pragma unroll
        for (int h = 0; h < 8; ++h) sm += o[h][r];
        float mu = redux16(sm) * (1.f / 128.f);
        float vs = 0.f;
        #pragma unroll
        for (int h = 0; h < 8; ++h) { float d = o[h][r] - mu; vs += d * d; }
        float var  = redux16(vs) * (1.f / 128.f);
        float rstd = rsqrtf(var + 1e-5f);
        int m = r + hi * 8;
        #pragma unroll
        for (int h = 0; h < 8; ++h) {
            int c = h * 16 + lo;
            out[(size_t)(a0 + m) * F_DIM + c] = (o[h][r] - mu) * rstd * gamma[c] + beta[c];
        }
    }
}

// ---------------- Launch ----------------
extern "C" void kernel_launch(void* const* d_in, const int* in_sizes, int n_in,
                              void* d_out, int out_size, void* d_ws, size_t ws_size,
                              hipStream_t stream) {
    const float*     atom    = (const float*)d_in[0];
    const float*     nbr_fea = (const float*)d_in[1];
    const long long* nbr_idx = (const long long*)d_in[2];   // int64
    const float* Wq = (const float*)d_in[3];  const float* bq = (const float*)d_in[4];
    const float* Wk = (const float*)d_in[5];  const float* bk = (const float*)d_in[6];
    const float* Wv = (const float*)d_in[7];  const float* bv = (const float*)d_in[8];
    const float* Wn = (const float*)d_in[9];  const float* bn = (const float*)d_in[10];
    const float* Wo = (const float*)d_in[11]; const float* bo = (const float*)d_in[12];
    const float* Wg = (const float*)d_in[13]; const float* bg = (const float*)d_in[14];
    const float* gamma = (const float*)d_in[15];
    const float* beta  = (const float*)d_in[16];

    // workspace layout: bf16 weights (transposed), then q/k/v fp32
    unsigned short* wbf = (unsigned short*)d_ws;
    unsigned short* wqt = wbf + 0 * 16384;
    unsigned short* wkt = wbf + 1 * 16384;
    unsigned short* wvt = wbf + 2 * 16384;
    unsigned short* wot = wbf + 3 * 16384;
    unsigned short* wnt = wbf + 4 * 16384;                   // 128x64
    float* qb = (float*)((char*)d_ws + 147456);
    float* kb = qb + (size_t)N_ATOMS * F_DIM;
    float* vb = kb + (size_t)N_ATOMS * F_DIM;

    prep_weights_kernel<<<72, 256, 0, stream>>>(Wq, Wk, Wv, Wo, Wn, wbf);
    qkv_kernel<<<N_TILES, 32, 0, stream>>>(atom, bq, bk, bv, wqt, wkt, wvt, qb, kb, vb);
    attn_kernel<<<N_TILES, 32, 0, stream>>>(atom, nbr_fea, nbr_idx, wnt, wot, bn, bo,
                                            Wg, bg, gamma, beta, qb, kb, vb, (float*)d_out);
}
